// DPSR_38637525795469
// MI455X (gfx1250) — compile-verified, matrix operand order checked by reading, not run on hardware
//
#include <hip/hip_runtime.h>
#include <stdint.h>

// ---------------------------------------------------------------------------
// Problem constants (match reference)
// ---------------------------------------------------------------------------
#define BB   64
#define TT   50
#define EE   1024
#define HH   1024
#define NI   32000
#define KBUF 4
#define WWIN 32
#define NF   10
#define FUL  (NF + 2*HH)          // 2058
#define FULP 2080                 // K padded to multiple of 32
#define LOUT (HH - WWIN + 1)      // 993
#define ROWS (BB*TT)              // 3200

typedef unsigned short ushort_t;
typedef __attribute__((ext_vector_type(8)))  __bf16 v8bf;
typedef __attribute__((ext_vector_type(16))) __bf16 v16bf;
typedef __attribute__((ext_vector_type(8)))  float  v8f;

#define DEV __device__ __forceinline__

DEV ushort_t f2bf(float f) {
  union { float f; unsigned u; } c; c.f = f;
  unsigned u = c.u;
  unsigned r = (u + 0x7FFFu + ((u >> 16) & 1u)) >> 16;   // RNE
  return (ushort_t)r;
}
DEV float bf2f(ushort_t h) {
  union { unsigned u; float f; } c; c.u = ((unsigned)h) << 16;
  return c.f;
}
DEV float sigm(float x) { return 1.f / (1.f + __expf(-x)); }
DEV v16bf cat16(v8bf lo, v8bf hi) {
  return __builtin_shufflevector(lo, hi, 0,1,2,3,4,5,6,7,8,9,10,11,12,13,14,15);
}

// ---------------------------------------------------------------------------
// Generic bf16 WMMA GEMM (one 16x16 tile / wave) — used for the small
// latency-bound per-step GEMMs (M=64) where occupancy matters most.
// C[MxN](f32) = A[MxK](bf16 row-major, lda) x Bt[NxK](bf16 row-major, ldb)^T
// grid = (Ntiles/8, Mtiles); K % 32 == 0; bias fused (nullable).
// ---------------------------------------------------------------------------
__global__ __launch_bounds__(256)
void gemm_bf16_wmma(const ushort_t* __restrict__ A, int lda,
                    const ushort_t* __restrict__ Bt, int ldb,
                    float* __restrict__ C, int ldc, int K,
                    const float* __restrict__ bias)
{
  const int wave = threadIdx.x >> 5;
  const int lane = threadIdx.x & 31;
  const int l16  = lane & 15;
  const int hi   = lane >> 4;                 // K-half select (0 or 8)
  const int tn   = blockIdx.x * 8 + wave;
  const int tm   = blockIdx.y;

  const ushort_t* Ap = A  + (size_t)(tm*16 + l16) * lda + hi*8;
  const ushort_t* Bp = Bt + (size_t)(tn*16 + l16) * ldb + hi*8;

  v8f acc = {};
  for (int kk = 0; kk < K; kk += 32) {
    v8bf a0 = *(const v8bf*)(Ap + kk);
    v8bf a1 = *(const v8bf*)(Ap + kk + 16);
    v8bf b0 = *(const v8bf*)(Bp + kk);
    v8bf b1 = *(const v8bf*)(Bp + kk + 16);
    acc = __builtin_amdgcn_wmma_f32_16x16x32_bf16(false, cat16(a0, a1),
                                                  false, cat16(b0, b1),
                                                  (short)0, acc, false, false);
  }

  const int col  = tn*16 + l16;
  const int row0 = tm*16 + hi*8;
  const float bv = bias ? bias[col] : 0.f;
#pragma unroll
  for (int r = 0; r < 8; ++r)
    C[(size_t)(row0 + r) * ldc + col] = acc[r] + bv;
}

// ---------------------------------------------------------------------------
// Register-blocked bf16 WMMA GEMM for the big bandwidth/compute-heavy GEMMs.
// Block tile 64(M) x 128(N); 8 waves as 2(M) x 4(N); each wave owns a 32x32
// output (4 accumulators) -> 4 WMMAs per 8 fragment loads.
// The shared 64x32 A panel is staged in LDS via ASYNC copies
// (global_load_async_to_lds_b128, ASYNCcnt) with double buffering; waves
// read A fragments with ds_load_b128. B streams from global (L2-resident).
// grid = (N/128, M/64); M % 64 == 0, N % 128 == 0, K % 32 == 0.
// ---------------------------------------------------------------------------
__global__ __launch_bounds__(256)
void gemm_bf16_wmma_big(const ushort_t* __restrict__ A, int lda,
                        const ushort_t* __restrict__ Bt, int ldb,
                        float* __restrict__ C, int ldc, int K,
                        const float* __restrict__ bias)
{
  __shared__ ushort_t sA[2][64 * 32];          // 2 x 4 KB double buffer

  const int tid  = threadIdx.x;
  const int wave = tid >> 5;
  const int lane = tid & 31;
  const int l16  = lane & 15;
  const int hi   = lane >> 4;
  const int wm   = wave >> 2;                  // 0..1  -> M offset wm*32
  const int wn   = wave & 3;                   // 0..3  -> N offset wn*32

  const int rowBase = blockIdx.y * 64;
  const int colBase = blockIdx.x * 128;

  // Async-copy mapping: 256 threads x 16B == one 64x32 bf16 A panel.
  const int ar = tid >> 2;                     // 0..63 (panel row)
  const int ac = (tid & 3) * 8;                // 0,8,16,24 (panel col, elems)
  const ushort_t* Ag = A + (size_t)(rowBase + ar) * lda + ac;
  // Generic->u32 truncation == LDS byte address (flat aperture rule).
  const unsigned lds0 = (unsigned)(uintptr_t)&sA[0][ar * 32 + ac];
  const unsigned lds1 = (unsigned)(uintptr_t)&sA[1][ar * 32 + ac];

  // Per-lane B fragment bases for the two N sub-tiles.
  const ushort_t* Bp0 = Bt + (size_t)(colBase + wn*32 + l16) * ldb + hi*8;
  const ushort_t* Bp1 = Bp0 + (size_t)16 * ldb;

  // Per-lane LDS read offsets (elements) for the two M sub-tiles.
  const int offA0 = (wm*32 + l16) * 32 + hi*8;
  const int offA1 = offA0 + 16 * 32;

  v8f acc00 = {}, acc01 = {}, acc10 = {}, acc11 = {};

  // Prime buffer 0.
  {
    unsigned long long ga = (unsigned long long)(uintptr_t)Ag;
    asm volatile("global_load_async_to_lds_b128 %0, %1, off"
                 :: "v"(lds0), "v"(ga) : "memory");
  }

  for (int kk = 0; kk < K; kk += 32) {
    // Complete this wave's pending fill; barrier => whole panel resident.
    asm volatile("s_wait_asynccnt 0x0" ::: "memory");
    __syncthreads();

    const int buf = (kk >> 5) & 1;
    if (kk + 32 < K) {                          // prefetch next panel
      unsigned long long ga = (unsigned long long)(uintptr_t)(Ag + kk + 32);
      unsigned ldsn = buf ? lds0 : lds1;
      asm volatile("global_load_async_to_lds_b128 %0, %1, off"
                   :: "v"(ldsn), "v"(ga) : "memory");
    }

    const ushort_t* sAb = &sA[buf][0];
    v8bf a00 = *(const v8bf*)(sAb + offA0);
    v8bf a01 = *(const v8bf*)(sAb + offA0 + 16);
    v8bf a10 = *(const v8bf*)(sAb + offA1);
    v8bf a11 = *(const v8bf*)(sAb + offA1 + 16);
    v8bf b00 = *(const v8bf*)(Bp0 + kk);
    v8bf b01 = *(const v8bf*)(Bp0 + kk + 16);
    v8bf b10 = *(const v8bf*)(Bp1 + kk);
    v8bf b11 = *(const v8bf*)(Bp1 + kk + 16);

    v16bf a0 = cat16(a00, a01), a1 = cat16(a10, a11);
    v16bf b0 = cat16(b00, b01), b1 = cat16(b10, b11);

    acc00 = __builtin_amdgcn_wmma_f32_16x16x32_bf16(false, a0, false, b0,
                                                    (short)0, acc00, false, false);
    acc01 = __builtin_amdgcn_wmma_f32_16x16x32_bf16(false, a0, false, b1,
                                                    (short)0, acc01, false, false);
    acc10 = __builtin_amdgcn_wmma_f32_16x16x32_bf16(false, a1, false, b0,
                                                    (short)0, acc10, false, false);
    acc11 = __builtin_amdgcn_wmma_f32_16x16x32_bf16(false, a1, false, b1,
                                                    (short)0, acc11, false, false);
  }

  // Epilogue: four 16x16 tiles.
  const int colA = colBase + wn*32 + l16;
  const int colB = colA + 16;
  const int rowA = rowBase + wm*32 + hi*8;
  const int rowB = rowA + 16;
  const float bvA = bias ? bias[colA] : 0.f;
  const float bvB = bias ? bias[colB] : 0.f;
#pragma unroll
  for (int r = 0; r < 8; ++r) {
    C[(size_t)(rowA + r) * ldc + colA] = acc00[r] + bvA;
    C[(size_t)(rowA + r) * ldc + colB] = acc01[r] + bvB;
    C[(size_t)(rowB + r) * ldc + colA] = acc10[r] + bvA;
    C[(size_t)(rowB + r) * ldc + colB] = acc11[r] + bvB;
  }
}

// ---------------------------------------------------------------------------
// fp32 -> bf16 flat convert (W_ih / W_hh are already [n][k])
// ---------------------------------------------------------------------------
__global__ void conv_bf16_flat(const float* __restrict__ in,
                               ushort_t* __restrict__ out, int n)
{
  int i0 = (blockIdx.x * 256 + threadIdx.x) * 4;
#pragma unroll
  for (int c = 0; c < 4; ++c) {
    int i = i0 + c;
    if (i < n) out[i] = f2bf(in[i]);
  }
}

// ---------------------------------------------------------------------------
// Tiled transpose + convert: out[c*out_ld + r] = bf16(in[r*in_cols + c])
// ---------------------------------------------------------------------------
__global__ void transpose_to_bf16(const float* __restrict__ in, int in_rows,
                                  int in_cols, ushort_t* __restrict__ out,
                                  int out_ld)
{
  __shared__ float tile[32][33];
  int c0 = blockIdx.x * 32, r0 = blockIdx.y * 32;
  for (int i = threadIdx.y; i < 32; i += 8) {
    int r = r0 + i, c = c0 + threadIdx.x;
    tile[i][threadIdx.x] = (r < in_rows && c < in_cols)
                               ? in[(size_t)r * in_cols + c] : 0.f;
  }
  __syncthreads();
  for (int i = threadIdx.y; i < 32; i += 8) {
    int c = c0 + i, r = r0 + threadIdx.x;
    if (c < in_cols && r < in_rows)
      out[(size_t)c * out_ld + r] = f2bf(tile[threadIdx.x][i]);
  }
}

// ---------------------------------------------------------------------------
// Embedding gather -> concatenated bf16 [u | it] rows of width 2E
// ---------------------------------------------------------------------------
__global__ void gather_emb(const int* __restrict__ uidx,
                           const int* __restrict__ iidx,
                           const float* __restrict__ uemb,
                           const float* __restrict__ iemb,
                           ushort_t* __restrict__ ui_b)
{
  int row = blockIdx.x;                 // 0..ROWS-1  (= b*T + t)
  int ur = uidx[row], ir = iidx[row];
#pragma unroll
  for (int q = 0; q < 4; ++q) {
    int e = threadIdx.x + 256 * q;
    ui_b[(size_t)row * (2*EE) + e]      = f2bf(uemb[(size_t)ur * EE + e]);
    ui_b[(size_t)row * (2*EE) + EE + e] = f2bf(iemb[(size_t)ir * EE + e]);
  }
}

// ---------------------------------------------------------------------------
// Per-step: att = sigmoid(att_pre(+att_b) + h@Wh); x = [att*u, (1-att)*it]
// ---------------------------------------------------------------------------
__global__ void att_x_kernel(const float* __restrict__ att_pre,   // (ROWS,E)
                             const float* __restrict__ hWh,       // (B,E)
                             const ushort_t* __restrict__ ui_b,   // (ROWS,2E)
                             ushort_t* __restrict__ x_b,          // (B,2E)
                             int t)
{
  int b = blockIdx.x;
  size_t row = (size_t)b * TT + t;
#pragma unroll
  for (int q = 0; q < 4; ++q) {
    int e = threadIdx.x + 256 * q;
    float a   = att_pre[row * EE + e] + hWh[(size_t)b * EE + e];
    float att = sigm(a);
    float uu  = bf2f(ui_b[row * (2*EE) + e]);
    float ii  = bf2f(ui_b[row * (2*EE) + EE + e]);
    x_b[(size_t)b * (2*EE) + e]      = f2bf(att * uu);
    x_b[(size_t)b * (2*EE) + EE + e] = f2bf((1.f - att) * ii);
  }
}

// ---------------------------------------------------------------------------
// Per-step GRU update + window-sums v + ring q/l, writes ful (f32 + bf16)
// ---------------------------------------------------------------------------
__global__ void gru_post(const float* __restrict__ gi,   // (B,3H) incl b_ih
                         const float* __restrict__ gh,   // (B,3H) incl b_hh
                         float* __restrict__ h_cur,      // (B,H)
                         ushort_t* __restrict__ h_b,     // (B,H)
                         float* __restrict__ ful_f,      // (ROWS,FUL)
                         ushort_t* __restrict__ ful_b,   // (ROWS,FULP)
                         const float* __restrict__ hcov_W,  // (NF,WWIN)
                         const float* __restrict__ hcov_b,  // (NF)
                         const float* __restrict__ vcov_W,  // (KBUF)
                         const float* __restrict__ vcov_b,  // (1)
                         int t)
{
  __shared__ float csum[HH];
  __shared__ float part[256];
  __shared__ float cs[WWIN];

  int b = blockIdx.x, j = threadIdx.x;
  size_t gb = (size_t)b * (3*HH);
  size_t row = (size_t)b * TT + t;

  float hn[4];
#pragma unroll
  for (int c = 0; c < 4; ++c) {
    int e = j * 4 + c;
    float r  = sigm(gi[gb + e]          + gh[gb + e]);
    float z  = sigm(gi[gb + HH + e]     + gh[gb + HH + e]);
    float n_ = tanhf(gi[gb + 2*HH + e]  + r * gh[gb + 2*HH + e]);
    float hp = h_cur[(size_t)b * HH + e];
    hn[c] = (1.f - z) * n_ + z * hp;
  }

  // --- block-wide prefix sum over the 1024 h values ---
  part[j] = hn[0] + hn[1] + hn[2] + hn[3];
  __syncthreads();
  if (j == 0) {
    float run = 0.f;
    for (int i = 0; i < 256; ++i) { float v = part[i]; part[i] = run; run += v; }
  }
  __syncthreads();
  {
    float run = part[j];
#pragma unroll
    for (int c = 0; c < 4; ++c) { run += hn[c]; csum[j*4 + c] = run; }
  }
  __syncthreads();

  if (j < WWIN) cs[j] = csum[LOUT - 1 + j] - (j > 0 ? csum[j - 1] : 0.f);
  __syncthreads();

  if (j < NF) {
    float v = hcov_b[j] * (float)LOUT;
#pragma unroll
    for (int w = 0; w < WWIN; ++w) v += hcov_W[j * WWIN + w] * cs[w];
    ful_f[row * FUL + j] = v;
    ful_b[row * FULP + j] = f2bf(v);
  }

  // --- q / l from the last K hidden states (read from stored h history) ---
  float w0 = vcov_W[0], w1 = vcov_W[1], w2 = vcov_W[2], w3 = vcov_W[3];
  float vb = vcov_b[0];
#pragma unroll
  for (int c = 0; c < 4; ++c) {
    int e = j * 4 + c;
    float h = hn[c];
    float l = 0.f;
    if (t >= KBUF - 1) {
      float q = vb + w3 * h;
      q += w0 * ful_f[((size_t)b * TT + (t - 3)) * FUL + NF + e];
      q += w1 * ful_f[((size_t)b * TT + (t - 2)) * FUL + NF + e];
      q += w2 * ful_f[((size_t)b * TT + (t - 1)) * FUL + NF + e];
      l = h * q;
    }
    ful_f[row * FUL + NF + e]      = h;
    ful_f[row * FUL + NF + HH + e] = l;
    ful_b[row * FULP + NF + e]      = f2bf(h);
    ful_b[row * FULP + NF + HH + e] = f2bf(l);
    h_cur[(size_t)b * HH + e] = h;
    h_b[(size_t)b * HH + e]   = f2bf(h);
  }
}

// ---------------------------------------------------------------------------
// In-place log-softmax over the class dim; one block per (b,t) row
// ---------------------------------------------------------------------------
__global__ void log_softmax_rows(float* __restrict__ out)
{
  __shared__ float red[256];
  size_t base = (size_t)blockIdx.x * NI;
  int j = threadIdx.x;

  float m = -3.4e38f;
  for (int i = j; i < NI; i += 256) m = fmaxf(m, out[base + i]);
  red[j] = m; __syncthreads();
  for (int s = 128; s > 0; s >>= 1) {
    if (j < s) red[j] = fmaxf(red[j], red[j + s]);
    __syncthreads();
  }
  m = red[0]; __syncthreads();

  float sum = 0.f;
  for (int i = j; i < NI; i += 256) sum += __expf(out[base + i] - m);
  red[j] = sum; __syncthreads();
  for (int s = 128; s > 0; s >>= 1) {
    if (j < s) red[j] += red[j + s];
    __syncthreads();
  }
  float lse = m + __logf(red[0]);
  __syncthreads();

  for (int i = j; i < NI; i += 256) out[base + i] -= lse;
}

// ---------------------------------------------------------------------------
// Host orchestration
// ---------------------------------------------------------------------------
extern "C" void kernel_launch(void* const* d_in, const int* in_sizes, int n_in,
                              void* d_out, int out_size, void* d_ws, size_t ws_size,
                              hipStream_t stream)
{
  const int*   user_vectors = (const int*)  d_in[0];
  const int*   item_vectors = (const int*)  d_in[1];
  const float* user_emb     = (const float*)d_in[2];
  const float* item_emb     = (const float*)d_in[3];
  const float* att_W        = (const float*)d_in[4];   // (3072,1024)
  const float* att_b        = (const float*)d_in[5];
  const float* W_ih         = (const float*)d_in[6];   // (3072,2048)
  const float* b_ih         = (const float*)d_in[7];
  const float* W_hh         = (const float*)d_in[8];   // (3072,1024)
  const float* b_hh         = (const float*)d_in[9];
  const float* hcov_W       = (const float*)d_in[10];
  const float* hcov_b       = (const float*)d_in[11];
  const float* vcov_W       = (const float*)d_in[12];
  const float* vcov_b       = (const float*)d_in[13];
  const float* lin_W        = (const float*)d_in[14];  // (FUL, NI)
  const float* lin_b        = (const float*)d_in[15];

  // ---- carve workspace ----
  char* p = (char*)d_ws;
  auto carve = [&](size_t bytes) -> void* {
    void* r = (void*)p;
    p += (bytes + 255) & ~(size_t)255;
    return r;
  };
  ushort_t* ui_b   = (ushort_t*)carve((size_t)ROWS * 2*EE * 2);
  ushort_t* attT   = (ushort_t*)carve((size_t)EE * (2*EE + HH) * 2);
  ushort_t* Wih_b  = (ushort_t*)carve((size_t)3*HH * 2*EE * 2);
  ushort_t* Whh_b  = (ushort_t*)carve((size_t)3*HH * HH * 2);
  ushort_t* linT   = (ushort_t*)carve((size_t)NI * FULP * 2);
  float*    attpre = (float*)   carve((size_t)ROWS * EE * 4);
  float*    ful_f  = (float*)   carve((size_t)ROWS * FUL * 4);
  ushort_t* ful_b  = (ushort_t*)carve((size_t)ROWS * FULP * 2);
  float*    h_cur  = (float*)   carve((size_t)BB * HH * 4);
  ushort_t* h_b    = (ushort_t*)carve((size_t)BB * HH * 2);
  float*    hWh    = (float*)   carve((size_t)BB * HH * 4);
  ushort_t* x_b    = (ushort_t*)carve((size_t)BB * 2*EE * 2);
  float*    gi     = (float*)   carve((size_t)BB * 3*HH * 4);
  float*    gh     = (float*)   carve((size_t)BB * 3*HH * 4);

  float* logits = (float*)d_out;   // (ROWS, NI), log-softmax in place

  // ---- zero-init state & pad regions (every call: deterministic) ----
  hipMemsetAsync(h_cur, 0, (size_t)BB * HH * 4, stream);
  hipMemsetAsync(h_b,   0, (size_t)BB * HH * 2, stream);
  hipMemsetAsync(ful_b, 0, (size_t)ROWS * FULP * 2, stream);
  hipMemsetAsync(linT,  0, (size_t)NI * FULP * 2, stream);

  // ---- weight conversion / transposition ----
  transpose_to_bf16<<<dim3((EE+31)/32, (2*EE+HH+31)/32), dim3(32,8), 0, stream>>>(
      att_W, 2*EE + HH, EE, attT, 2*EE + HH);
  conv_bf16_flat<<<(3*HH*2*EE + 1023)/1024, 256, 0, stream>>>(W_ih, Wih_b, 3*HH*2*EE);
  conv_bf16_flat<<<(3*HH*HH   + 1023)/1024, 256, 0, stream>>>(W_hh, Whh_b, 3*HH*HH);
  transpose_to_bf16<<<dim3((NI+31)/32, (FUL+31)/32), dim3(32,8), 0, stream>>>(
      lin_W, FUL, NI, linT, FULP);

  // ---- embedding gather ----
  gather_emb<<<ROWS, 256, 0, stream>>>(user_vectors, item_vectors,
                                       user_emb, item_emb, ui_b);

  // ---- att_pre = [u|it] @ [Wu;Wi] + att_b  (M=3200,K=2048,N=1024) ----
  gemm_bf16_wmma_big<<<dim3(EE/128, ROWS/64), 256, 0, stream>>>(
      ui_b, 2*EE, attT, 2*EE + HH, attpre, EE, 2*EE, att_b);

  // ---- sequential attention-GRU scan ----
  for (int t = 0; t < TT; ++t) {
    gemm_bf16_wmma<<<dim3(EE/16/8, BB/16), 256, 0, stream>>>(
        h_b, HH, attT + 2*EE, 2*EE + HH, hWh, EE, HH, nullptr);

    att_x_kernel<<<BB, 256, 0, stream>>>(attpre, hWh, ui_b, x_b, t);

    gemm_bf16_wmma<<<dim3(3*HH/16/8, BB/16), 256, 0, stream>>>(
        x_b, 2*EE, Wih_b, 2*EE, gi, 3*HH, 2*EE, b_ih);

    gemm_bf16_wmma<<<dim3(3*HH/16/8, BB/16), 256, 0, stream>>>(
        h_b, HH, Whh_b, HH, gh, 3*HH, HH, b_hh);

    gru_post<<<BB, 256, 0, stream>>>(gi, gh, h_cur, h_b, ful_f, ful_b,
                                     hcov_W, hcov_b, vcov_W, vcov_b, t);
  }

  // ---- logits = ful @ lin_W + lin_b  (M=3200, K=2080, N=32000) ----
  gemm_bf16_wmma_big<<<dim3(NI/128, ROWS/64), 256, 0, stream>>>(
      ful_b, FULP, linT, FULP, logits, NI, FULP, lin_b);

  // ---- log-softmax in place ----
  log_softmax_rows<<<ROWS, 256, 0, stream>>>(logits);
}
